// HSTULayer_44195213476040
// MI455X (gfx1250) — matrix operand order repaired and checked by reading, not compiled, not run on hardware
//
#include <hip/hip_runtime.h>

// Problem constants (match reference)
#define Bx   2
#define Tt   2048
#define Dd   512
#define Hh   8
#define HDm  64
#define FFm  2048
#define MPm  2048
#define Mrows (Bx * Tt)   // 4096

typedef __attribute__((ext_vector_type(16))) _Float16 v16h;
typedef __attribute__((ext_vector_type(8)))  float    v8f;
typedef __attribute__((ext_vector_type(4)))  int      v4i;

// ---- CDNA5 async global->LDS copy (ASYNCcnt path), with portable fallback ----
#if defined(__has_builtin)
#if __has_builtin(__builtin_amdgcn_global_load_async_to_lds_b128) && \
    __has_builtin(__builtin_amdgcn_s_wait_asynccnt)
#define USE_ASYNC_LDS 1
#endif
#endif
#ifndef USE_ASYNC_LDS
#define USE_ASYNC_LDS 0
#endif

__device__ __forceinline__ void cp_g2l_16B(const _Float16* __restrict__ g, _Float16* l) {
#if USE_ASYNC_LDS
  typedef __attribute__((address_space(1))) v4i gv4i;   // global int4
  typedef __attribute__((address_space(3))) v4i lv4i;   // LDS int4
  __builtin_amdgcn_global_load_async_to_lds_b128((gv4i*)g, (lv4i*)l, 0, 0);
#else
  *(uint4*)l = *(const uint4*)g;
#endif
}
__device__ __forceinline__ void cp_commit_wait() {
#if USE_ASYNC_LDS
  __builtin_amdgcn_s_wait_asynccnt(0);
#endif
  __syncthreads();
}

__device__ __forceinline__ v8f wmma16(v16h a, v16h b, v8f c) {
  // D = A(16x32 f16) * B(32x16 f16) + C(16x16 f32)
  return __builtin_amdgcn_wmma_f32_16x16x32_f16(false, a, false, b, (short)0, c,
                                                false, false);
}

// A-fragment (16x32 f16, row-major source, leading dim ld).
// Per ISA 7.12.2: lanes 0-15 row=lane hold K{0..7,16..23}; lanes 16-31 hold K{8..15,24..31}.
__device__ __forceinline__ v16h load_a_frag(const _Float16* p, int ld) {
  const int lane = threadIdx.x & 31;
  const _Float16* q = p + (lane & 15) * ld + ((lane >> 4) << 3);
  union { v16h v; uint4 u[2]; } r;
  r.u[0] = *(const uint4*)(q);        // K +0..+7
  r.u[1] = *(const uint4*)(q + 16);   // K +16..+23
  return r.v;
}

// B-fragment (32x16 f16). Source is "B^T" row-major [N,K] with leading dim ld,
// so lane n reads K halves: lanes 0-15 K{0..7,16..23}, lanes 16-31 K{8..15,24..31}.
__device__ __forceinline__ v16h load_b_frag(const _Float16* p, int ld) {
  const int lane = threadIdx.x & 31;
  const _Float16* q = p + (lane & 15) * ld + ((lane >> 4) << 3);
  union { v16h v; uint4 u[2]; } r;
  r.u[0] = *(const uint4*)(q);
  r.u[1] = *(const uint4*)(q + 16);
  return r.v;
}

// ---------------- LayerNorm (one wave32 per row of D=512), f16 output ----------------
__global__ void __launch_bounds__(256)
ln_f16_kernel(const float* __restrict__ x, const float* __restrict__ g,
              const float* __restrict__ bta, _Float16* __restrict__ y) {
  const int lane = threadIdx.x & 31;
  const int row  = blockIdx.x * 8 + (threadIdx.x >> 5);
  const float* xr = x + (size_t)row * Dd;
  float v[16];
  float s = 0.f;
#pragma unroll
  for (int i = 0; i < 16; ++i) { v[i] = xr[lane + i * 32]; s += v[i]; }
#pragma unroll
  for (int m = 1; m < 32; m <<= 1) s += __shfl_xor(s, m);
  const float mean = s * (1.f / Dd);
  float vv = 0.f;
#pragma unroll
  for (int i = 0; i < 16; ++i) { const float d = v[i] - mean; vv += d * d; }
#pragma unroll
  for (int m = 1; m < 32; m <<= 1) vv += __shfl_xor(vv, m);
  const float rs = rsqrtf(vv * (1.f / Dd) + 1e-5f);
#pragma unroll
  for (int i = 0; i < 16; ++i) {
    const int c = lane + i * 32;
    y[(size_t)row * Dd + c] = (_Float16)((v[i] - mean) * rs * g[c] + bta[c]);
  }
}

// ---------------- Weight cast + transpose: W[K,N] f32 -> Wt[N,K] f16 ----------------
__global__ void __launch_bounds__(256)
wcast_t_kernel(const float* __restrict__ W, _Float16* __restrict__ Wt, int K, int N) {
  const int idx = blockIdx.x * 256 + threadIdx.x;
  if (idx >= K * N) return;
  const int k = idx % K, n = idx / K;
  Wt[idx] = (_Float16)W[(size_t)k * N + n];
}

// ---------------- Templated WMMA GEMM ----------------
// C[M,N] = A[M,K](f16) * Bt[N,K]^T(f16) + bias (+resid) (relu?) -> f32 and/or f16.
// Block: 256 thr = 8 waves, tile 128(M) x 64(N); wave tile 32x32 (2x2 WMMA frags).
// A/B stream through L2 (192 MB -> fully resident), so direct fragment loads.
template <int RELU, int RESID, int OF32, int OF16>
__global__ void __launch_bounds__(256)
gemm_wmma_kernel(const _Float16* __restrict__ A, const _Float16* __restrict__ Bt,
                 const float* __restrict__ bias, const float* __restrict__ resid,
                 float* __restrict__ Cf, _Float16* __restrict__ Ch, int N, int K) {
  const int wave   = threadIdx.x >> 5;
  const int m_base = blockIdx.y * 128 + (wave >> 1) * 32;
  const int n_base = blockIdx.x * 64  + (wave & 1) * 32;
  v8f acc00 = {}, acc01 = {}, acc10 = {}, acc11 = {};
  const _Float16* Ar0 = A  + (size_t)m_base * K;
  const _Float16* Ar1 = A  + (size_t)(m_base + 16) * K;
  const _Float16* Br0 = Bt + (size_t)n_base * K;
  const _Float16* Br1 = Bt + (size_t)(n_base + 16) * K;
  for (int k0 = 0; k0 < K; k0 += 32) {
    v16h a0 = load_a_frag(Ar0 + k0, K);
    v16h a1 = load_a_frag(Ar1 + k0, K);
    v16h b0 = load_b_frag(Br0 + k0, K);
    v16h b1 = load_b_frag(Br1 + k0, K);
    acc00 = wmma16(a0, b0, acc00);
    acc01 = wmma16(a0, b1, acc01);
    acc10 = wmma16(a1, b0, acc10);
    acc11 = wmma16(a1, b1, acc11);
    if (k0 + 128 < K) {
      __builtin_prefetch(Ar0 + k0 + 128, 0, 1);   // global_prefetch_b8
      __builtin_prefetch(Br0 + k0 + 128, 0, 1);
    }
  }
  const int lane = threadIdx.x & 31;
  const int nn = lane & 15;
  const int mh = (lane >> 4) * 8;
  v8f* accs[4] = {&acc00, &acc01, &acc10, &acc11};
#pragma unroll
  for (int t = 0; t < 4; ++t) {
    const int i = t >> 1, j = t & 1;
    const int n = n_base + j * 16 + nn;
    const float bv = bias[n];
#pragma unroll
    for (int r = 0; r < 8; ++r) {
      const int m = m_base + i * 16 + mh + r;
      float v = (*accs[t])[r] + bv;
      if (RESID) v += resid[(size_t)m * N + n];
      if (RELU)  v = fmaxf(v, 0.f);
      if (OF32)  Cf[(size_t)m * N + n] = v;
      if (OF16)  Ch[(size_t)m * N + n] = (_Float16)v;
    }
  }
}

// ---------------- QKV reorg: qkv[M, 3D] f16 -> Qh/Kh [B,H,T,64], VhT [B,H,64,T] ----------------
__global__ void __launch_bounds__(256)
qkv_reorg_kernel(const _Float16* __restrict__ qkv, _Float16* __restrict__ Qh,
                 _Float16* __restrict__ Kh, _Float16* __restrict__ VhT) {
  const int idx = blockIdx.x * 256 + threadIdx.x;  // over B*T*H*HD = 2,097,152
  const int d = idx & 63;
  int t0 = idx >> 6;
  const int h = t0 & 7;  t0 >>= 3;
  const int t = t0 & (Tt - 1);
  const int b = t0 >> 11;
  const size_t row = (size_t)(b * Tt + t) * (3 * Dd);
  const int col = h * HDm + d;
  Qh [((size_t)((b * Hh + h) * Tt + t)) * HDm + d] = qkv[row + col];
  Kh [((size_t)((b * Hh + h) * Tt + t)) * HDm + d] = qkv[row + Dd + col];
  VhT[((size_t)((b * Hh + h) * HDm + d)) * Tt + t] = qkv[row + 2 * Dd + col];
}

// ---------------- Flash attention ----------------
// Block = 256 thr (8 waves); each wave owns 16 query rows; block tile = 128 q rows.
// Grid = B*H*(T/128) = 256 blocks. K/V tiles cooperatively staged into LDS with
// the CDNA5 async-to-LDS path (double-buffered); per 32-key block: 8 WMMA/wave.
__global__ void __launch_bounds__(256)
flash_attn_kernel(const _Float16* __restrict__ Qh, const _Float16* __restrict__ Kh,
                  const _Float16* __restrict__ VhT, const float* __restrict__ rel_table,
                  const float* __restrict__ mask, _Float16* __restrict__ attn_out) {
  __shared__ __align__(16) _Float16 kbuf[2][32 * HDm];  // [key 32][hd 64], 2x4 KB
  __shared__ __align__(16) _Float16 vbuf[2][HDm * 32];  // [hd 64][key 32], 2x4 KB
  __shared__ __align__(16) _Float16 pbuf[8][16 * 32];   // per-wave P tile, 8 KB
  const int tid  = threadIdx.x;
  const int lane = tid & 31;
  const int wave = tid >> 5;
  const int nn   = lane & 15;
  const int kg   = lane >> 4;

  int bid = blockIdx.x;
  const int qt = bid & 15; bid >>= 4;
  const int h  = bid & 7;  bid >>= 3;
  const int b  = bid;
  const int q0 = qt * 128 + wave * 16;

  const _Float16* Qb = Qh  + ((size_t)(b * Hh + h)) * Tt * HDm;
  const _Float16* Kb = Kh  + ((size_t)(b * Hh + h)) * Tt * HDm;
  const _Float16* Vb = VhT + ((size_t)(b * Hh + h)) * HDm * Tt;
  const float* mb = mask + (size_t)b * Tt * Tt;

  const v16h qa0 = load_a_frag(Qb + (size_t)q0 * HDm, HDm);
  const v16h qa1 = load_a_frag(Qb + (size_t)q0 * HDm + 32, HDm);

  // Cooperative stage of one 32-key K tile + V tile (256 thr x 16 B each side).
  auto stage = [&](int buf, int kb) {
    const int krow = tid >> 3, kc = (tid & 7) << 3;            // 32 rows x 64 halves
    cp_g2l_16B(Kb + (size_t)(kb + krow) * HDm + kc, &kbuf[buf][krow * HDm + kc]);
    const int vd = tid >> 2, vc = (tid & 3) << 3;              // 64 rows x 32 halves
    cp_g2l_16B(Vb + (size_t)vd * Tt + kb + vc, &vbuf[buf][vd * 32 + vc]);
  };

  float rmax[8], rsum[8];
  v8f o0 = {}, o1 = {}, o2 = {}, o3 = {};
  v8f* o[4] = {&o0, &o1, &o2, &o3};
#pragma unroll
  for (int r = 0; r < 8; ++r) { rmax[r] = -1e30f; rsum[r] = 0.f; }

  stage(0, 0);
  for (int kb = 0; kb < Tt; kb += 32) {
    const int buf = (kb >> 5) & 1;
    cp_commit_wait();                       // staged tile visible; prev compute done
    if (kb + 32 < Tt) stage(buf ^ 1, kb + 32);

    v8f s0 = {}, s1 = {};
    {
      v16h b0 = load_b_frag(&kbuf[buf][0], HDm);            // keys kb..+15, K 0..31
      v16h b1 = load_b_frag(&kbuf[buf][16 * HDm], HDm);     // keys kb+16..+31
      s0 = wmma16(qa0, b0, s0);
      s1 = wmma16(qa0, b1, s1);
      b0 = load_b_frag(&kbuf[buf][32], HDm);                // K 32..63
      b1 = load_b_frag(&kbuf[buf][16 * HDm + 32], HDm);
      s0 = wmma16(qa1, b0, s0);
      s1 = wmma16(qa1, b1, s1);
    }
    const int nk0 = kb + nn;
    const int nk1 = kb + 16 + nn;
#pragma unroll
    for (int r = 0; r < 8; ++r) {
      const int qrow = q0 + kg * 8 + r;
      int r0 = nk0 - qrow + (MPm - 1);
      int r1 = nk1 - qrow + (MPm - 1);
      r0 = r0 < 0 ? 0 : (r0 > 2 * MPm - 2 ? 2 * MPm - 2 : r0);
      r1 = r1 < 0 ? 0 : (r1 > 2 * MPm - 2 ? 2 * MPm - 2 : r1);
      s0[r] = s0[r] * 0.125f + rel_table[r0 * Hh + h] + mb[(size_t)qrow * Tt + nk0];
      s1[r] = s1[r] * 0.125f + rel_table[r1 * Hh + h] + mb[(size_t)qrow * Tt + nk1];
    }
    // Online softmax; row m = kg*8 + r lives in a fixed 16-lane half (xor 1/2/4/8).
#pragma unroll
    for (int r = 0; r < 8; ++r) {
      float v = fmaxf(s0[r], s1[r]);
      v = fmaxf(v, __shfl_xor(v, 1));
      v = fmaxf(v, __shfl_xor(v, 2));
      v = fmaxf(v, __shfl_xor(v, 4));
      v = fmaxf(v, __shfl_xor(v, 8));
      const float nm   = fmaxf(rmax[r], v);
      const float corr = __expf(rmax[r] - nm);
      rmax[r] = nm;
      const float p0 = __expf(s0[r] - nm);
      const float p1 = __expf(s1[r] - nm);
      float ps = p0 + p1;
      ps += __shfl_xor(ps, 1);
      ps += __shfl_xor(ps, 2);
      ps += __shfl_xor(ps, 4);
      ps += __shfl_xor(ps, 8);
      rsum[r] = rsum[r] * corr + ps;
#pragma unroll
      for (int j = 0; j < 4; ++j) (*o[j])[r] *= corr;
      const int m = kg * 8 + r;
      pbuf[wave][m * 32 + nn]      = (_Float16)p0;
      pbuf[wave][m * 32 + 16 + nn] = (_Float16)p1;
    }
    // P (16x32) from LDS as A-fragment; V columns as B-fragments from LDS V tile.
    const v16h pa = load_a_frag(&pbuf[wave][0], 32);
#pragma unroll
    for (int j = 0; j < 4; ++j) {
      v16h vb = load_b_frag(&vbuf[buf][(j * 16) * 32], 32);
      *o[j] = wmma16(pa, vb, *o[j]);
    }
  }
  // Epilogue: normalize and store f16 in [B,T,H*HD] layout for the out-proj GEMM.
#pragma unroll
  for (int j = 0; j < 4; ++j) {
#pragma unroll
    for (int r = 0; r < 8; ++r) {
      const int qrow = q0 + kg * 8 + r;
      const int d = j * 16 + nn;
      attn_out[((size_t)(b * Tt + qrow)) * Dd + h * HDm + d] =
          (_Float16)((*o[j])[r] / rsum[r]);
    }
  }
}

// ---------------- Host orchestration ----------------
extern "C" void kernel_launch(void* const* d_in, const int* in_sizes, int n_in,
                              void* d_out, int out_size, void* d_ws, size_t ws_size,
                              hipStream_t stream) {
  const float* x      = (const float*)d_in[0];
  const float* mask   = (const float*)d_in[1];
  const float* lp_w   = (const float*)d_in[2];
  const float* lp_b   = (const float*)d_in[3];
  const float* qkv_w  = (const float*)d_in[4];
  const float* qkv_b  = (const float*)d_in[5];
  const float* out_w  = (const float*)d_in[6];
  const float* out_b  = (const float*)d_in[7];
  const float* rel    = (const float*)d_in[8];
  const float* w1     = (const float*)d_in[9];
  const float* b1     = (const float*)d_in[10];
  const float* w2     = (const float*)d_in[11];
  const float* b2     = (const float*)d_in[12];
  const float* g1     = (const float*)d_in[13];
  const float* be1    = (const float*)d_in[14];
  const float* g2     = (const float*)d_in[15];
  const float* be2    = (const float*)d_in[16];
  const float* g3     = (const float*)d_in[17];
  const float* be3    = (const float*)d_in[18];

  char* wsp = (char*)d_ws;
  auto carve = [&](size_t bytes) -> char* {
    char* p = wsp;
    wsp += (bytes + 255) & ~(size_t)255;
    return p;
  };
  _Float16* lnbuf  = (_Float16*)carve((size_t)Mrows * Dd * 2);        // 4 MiB (reused 3x)
  float*    xproj  = (float*)   carve((size_t)Mrows * Dd * 4);        // 8 MiB
  _Float16* qkv    = (_Float16*)carve((size_t)Mrows * 3 * Dd * 2);    // 12 MiB
  _Float16* Qh     = (_Float16*)carve((size_t)Bx * Hh * Tt * HDm * 2);
  _Float16* Kh     = (_Float16*)carve((size_t)Bx * Hh * Tt * HDm * 2);
  _Float16* VhT    = (_Float16*)carve((size_t)Bx * Hh * HDm * Tt * 2);
  _Float16* attn   = (_Float16*)carve((size_t)Mrows * Dd * 2);
  float*    xattn  = (float*)   carve((size_t)Mrows * Dd * 4);
  _Float16* hbuf   = (_Float16*)carve((size_t)Mrows * FFm * 2);       // 16 MiB
  _Float16* lp_wT  = (_Float16*)carve((size_t)Dd * Dd * 2);
  _Float16* qkv_wT = (_Float16*)carve((size_t)Dd * 3 * Dd * 2);
  _Float16* out_wT = (_Float16*)carve((size_t)Dd * Dd * 2);
  _Float16* w1T    = (_Float16*)carve((size_t)Dd * FFm * 2);
  _Float16* w2T    = (_Float16*)carve((size_t)FFm * Dd * 2);

  // Weight transpose + f16 cast
  wcast_t_kernel<<<(Dd * Dd + 255) / 256, 256, 0, stream>>>(lp_w,  lp_wT,  Dd, Dd);
  wcast_t_kernel<<<(Dd * 3 * Dd + 255) / 256, 256, 0, stream>>>(qkv_w, qkv_wT, Dd, 3 * Dd);
  wcast_t_kernel<<<(Dd * Dd + 255) / 256, 256, 0, stream>>>(out_w, out_wT, Dd, Dd);
  wcast_t_kernel<<<(Dd * FFm + 255) / 256, 256, 0, stream>>>(w1, w1T, Dd, FFm);
  wcast_t_kernel<<<(FFm * Dd + 255) / 256, 256, 0, stream>>>(w2, w2T, FFm, Dd);

  // x_proj = LN(x;g1,be1) @ lp_w + lp_b
  ln_f16_kernel<<<Mrows / 8, 256, 0, stream>>>(x, g1, be1, lnbuf);
  gemm_wmma_kernel<0, 0, 1, 0><<<dim3(Dd / 64, Mrows / 128), 256, 0, stream>>>(
      lnbuf, lp_wT, lp_b, nullptr, xproj, nullptr, Dd, Dd);

  // qkv = LN(x_proj;g2,be2) @ qkv_w + qkv_b  (f16 out)
  ln_f16_kernel<<<Mrows / 8, 256, 0, stream>>>(xproj, g2, be2, lnbuf);
  gemm_wmma_kernel<0, 0, 0, 1><<<dim3(3 * Dd / 64, Mrows / 128), 256, 0, stream>>>(
      lnbuf, qkv_wT, qkv_b, nullptr, nullptr, qkv, 3 * Dd, Dd);

  // Split heads; V stored transposed for WMMA B-fragments
  qkv_reorg_kernel<<<(Bx * Tt * Hh * HDm) / 256, 256, 0, stream>>>(qkv, Qh, Kh, VhT);

  // Fused flash attention with rel-position bias + mask
  flash_attn_kernel<<<Bx * Hh * (Tt / 128), 256, 0, stream>>>(Qh, Kh, VhT, rel, mask, attn);

  // x_attn = x_proj + attn @ out_w + out_b
  gemm_wmma_kernel<0, 1, 1, 0><<<dim3(Dd / 64, Mrows / 128), 256, 0, stream>>>(
      attn, out_wT, out_b, xproj, xattn, nullptr, Dd, Dd);

  // h = relu(LN(x_attn;g3,be3) @ w1 + b1)  (f16 out)
  ln_f16_kernel<<<Mrows / 8, 256, 0, stream>>>(xattn, g3, be3, lnbuf);
  gemm_wmma_kernel<1, 0, 0, 1><<<dim3(FFm / 64, Mrows / 128), 256, 0, stream>>>(
      lnbuf, w1T, b1, nullptr, nullptr, hbuf, FFm, Dd);

  // out = x_attn + h @ w2 + b2
  gemm_wmma_kernel<0, 1, 1, 0><<<dim3(Dd / 64, Mrows / 128), 256, 0, stream>>>(
      hbuf, w2T, b2, xattn, (float*)d_out, nullptr, Dd, FFm);
}
